// interactionModule_nonParametric_acceleration_65549790871653
// MI455X (gfx1250) — matrix operand order repaired and checked by reading, not compiled, not run on hardware
//
#include <hip/hip_runtime.h>
#include <hip/hip_bf16.h>

// ---------------------------------------------------------------------------
// Fused edge-MLP force kernel for MI455X (gfx1250, wave32, WMMA).
//
//   per edge: adr = |x[dst]-x[src]|
//             h = relu(adr*W0+b0); h = relu(h@W1+b1); h = relu(h@W2+b2)
//             f = h@W3+b3 ;  out[dst] += f * unit_dr ;  out = sum - gamma*v
//
// One wave32 processes a tile of 16 edges through the whole MLP. The two
// 128x128 hidden layers use v_wmma_f32_16x16x32_f16 (8 N-tiles x 4 K-blocks
// = 32 WMMAs per layer). Activations live in a wave-private 16x128 f16 LDS
// tile; weights are pre-converted once to transposed f16 in d_ws.
//
// Anti-LICM trick: a per-iteration opaque *integer* offset (always 0) is
// added to the weight pointers. This keeps loads loop-variant (no hoist ->
// no spill-to-scratch as in round 1) while PRESERVING pointer provenance so
// the loads lower to global_load_b128 (not flat_load_b128 as in round 2,
// which double-counts into DScnt and burns 64-bit VGPR address math).
// ---------------------------------------------------------------------------

typedef __attribute__((ext_vector_type(16))) _Float16 v16h;
typedef __attribute__((ext_vector_type(8)))  _Float16 v8h;
typedef __attribute__((ext_vector_type(4)))  _Float16 v4h;
typedef __attribute__((ext_vector_type(8)))  float    v8f;
typedef __attribute__((ext_vector_type(4)))  float    v4f;

#define HIDDEN 128
#define TILE_M 16
#define WAVES_PER_BLOCK 8
#define BLOCK_THREADS (WAVES_PER_BLOCK * 32)
#define EDGE_BLOCKS 1024

union Frag16 {
    v16h v;
    v8h  h[2];
};

// --- weight pre-pass: f32 row-major W[k][n] -> f16 transposed Wt[n][k] -----
__global__ void wconvert_kernel(const float* __restrict__ W1,
                                const float* __restrict__ W2,
                                _Float16* __restrict__ ws) {
    int idx = blockIdx.x * blockDim.x + threadIdx.x;
    if (idx < HIDDEN * HIDDEN) {
        int k = idx >> 7;           // fan-in index
        int n = idx & (HIDDEN - 1); // fan-out index
        ws[n * HIDDEN + k]                   = (_Float16)W1[idx];
        ws[HIDDEN * HIDDEN + n * HIDDEN + k] = (_Float16)W2[idx];
    }
}

// --- output init: out = -gamma * v -----------------------------------------
__global__ void out_init_kernel(const float* __restrict__ v,
                                const float* __restrict__ gamma,
                                float* __restrict__ out, int n) {
    int i = blockIdx.x * blockDim.x + threadIdx.x;
    if (i < n) out[i] = -gamma[0] * v[i];
}

// --- one hidden layer: Anew(16x128) = relu(A(16x128) @ W(128x128) + b) -----
// A    : wave-private LDS tile, row-major f16, row stride HIDDEN
// Wt   : transposed f16 weights in global, Wt[n*128 + k] = W[k][n]
// bias : f32 in global
__device__ __forceinline__ void mlp_hidden_layer(_Float16* A,
                                                 const _Float16* __restrict__ Wt,
                                                 const float* __restrict__ bias,
                                                 int lane) {
    const int nl = lane & 15;   // column-within-tile / row index m
    const int hi = lane >> 4;   // half-wave select

    // Hoist the 4 K-block A fragments into registers (ISA 16-bit A layout:
    // lanes 0-15 hold K=[0..7]+[16..23] of row m=lane, lanes 16-31 hold
    // K=[8..15]+[24..31] of row m=lane-16 -> two contiguous 8-half runs).
    Frag16 af[4];
#pragma unroll
    for (int kb = 0; kb < 4; ++kb) {
        const _Float16* row = &A[nl * HIDDEN + kb * 32 + hi * 8];
        af[kb].h[0] = *(const v8h*)(row);        // K run 1 (16B -> ds b128)
        af[kb].h[1] = *(const v8h*)(row + 16);   // K run 2
    }

#pragma unroll
    for (int nt = 0; nt < 8; ++nt) {
        const int n0 = nt * 16;
        // C/D layout: element (M = r + 8*hi, N = n0 + nl); bias depends on N.
        const float bv = bias[n0 + nl];
        v8f acc = {bv, bv, bv, bv, bv, bv, bv, bv};
#pragma unroll
        for (int kb = 0; kb < 4; ++kb) {
            // B layout (16-bit, 32x16): same K order as A (lane group hi holds
            // K=[8hi..8hi+7]+[16+8hi..16+8hi+7]), lane&15 selects column N.
            const _Float16* bp = Wt + (n0 + nl) * HIDDEN + kb * 32 + hi * 8;
            Frag16 bf;
            bf.h[0] = *(const v8h*)(bp);         // global_load_b128 (WGP$ hot)
            bf.h[1] = *(const v8h*)(bp + 16);
            acc = __builtin_amdgcn_wmma_f32_16x16x32_f16(
                /*neg_a=*/false, af[kb].v, /*neg_b=*/false, bf.v,
                /*c_mod=*/(short)0, acc, /*reuse_a=*/false, /*reuse_b=*/false);
        }
        // ReLU in f32, convert, write back in place (A frags are in regs).
#pragma unroll
        for (int r = 0; r < 8; ++r) {
            float hv = fmaxf(acc[r], 0.0f);
            A[(r + 8 * hi) * HIDDEN + n0 + nl] = (_Float16)hv;
        }
    }
}

__launch_bounds__(BLOCK_THREADS, 1)
__global__ void edge_force_kernel(const float* __restrict__ x,
                                  const int* __restrict__ src,
                                  const int* __restrict__ dst,
                                  const float* __restrict__ W0g,
                                  const float* __restrict__ b0g,
                                  const float* __restrict__ b1,
                                  const float* __restrict__ b2,
                                  const float* __restrict__ W3g,
                                  const float* __restrict__ b3,
                                  const _Float16* __restrict__ Wt1g,
                                  const _Float16* __restrict__ Wt2g,
                                  float* __restrict__ out,
                                  int nEdges) {
    __shared__ __align__(16) _Float16 sA[WAVES_PER_BLOCK][TILE_M * HIDDEN];
    __shared__ float sAbs[WAVES_PER_BLOCK][TILE_M];
    __shared__ float sRed[WAVES_PER_BLOCK][32];

    const int lane = threadIdx.x & 31;
    const int wave = threadIdx.x >> 5;
    const int nl   = lane & 15;
    const int hi   = lane >> 4;

    _Float16* A = sA[wave];

    const int nTiles  = (nEdges + TILE_M - 1) / TILE_M;
    const int gw      = blockIdx.x * WAVES_PER_BLOCK + wave;
    const int strideW = gridDim.x * WAVES_PER_BLOCK;

    for (int tile = gw; tile < nTiles; tile += strideW) {
        // Opaque zero offset: loop-variant as far as the compiler knows, so
        // weight loads cannot be hoisted out of the tile loop, but pointer
        // provenance is preserved -> loads stay GLOBAL (not FLAT).
        int iv = 0;
        asm volatile("" : "+s"(iv));
        const float*    W0  = W0g  + iv;
        const float*    b0  = b0g  + iv;
        const float*    W3  = W3g  + iv;
        const _Float16* Wt1 = Wt1g + iv;
        const _Float16* Wt2 = Wt2g + iv;

        // ---- gather + distance (lanes 0..15 own one edge each) -----------
        float ux = 0.0f, uy = 0.0f;
        int   dIdx  = 0;
        bool  valid = false;
        if (hi == 0) {
            const int e = tile * TILE_M + nl;
            float adr = 0.0f;
            if (e < nEdges) {
                const int s = src[e];
                dIdx        = dst[e];
                const float dx = x[2 * dIdx]     - x[2 * s];
                const float dy = x[2 * dIdx + 1] - x[2 * s + 1];
                adr = sqrtf(dx * dx + dy * dy);
                const float inv = 1.0f / fmaxf(adr, 1e-12f);
                ux = dx * inv;
                uy = dy * inv;
                valid = true;
            }
            sAbs[wave][nl] = adr;   // broadcast through wave-private LDS
        }
        // same-wave LDS ops are in order; no barrier needed.
        const float adr_m = sAbs[wave][nl];

        // ---- layer 0: h0[k] = relu(adr*W0[k] + b0[k]) --------------------
        // lane covers row m = nl, K half hi*64 .. hi*64+63 (vectorized x4)
        {
            const int kbase = hi * 64;
#pragma unroll
            for (int j = 0; j < 64; j += 4) {
                const v4f w = *(const v4f*)(W0 + kbase + j);
                const v4f b = *(const v4f*)(b0 + kbase + j);
                v4h hh;
#pragma unroll
                for (int i = 0; i < 4; ++i) {
                    float h = fmaf(adr_m, w[i], b[i]);
                    hh[i] = (_Float16)fmaxf(h, 0.0f);
                }
                *(v4h*)&A[nl * HIDDEN + kbase + j] = hh;   // ds_store_b64
            }
        }

        // ---- layers 1 & 2: WMMA (uniform control flow, EXEC all-1s) ------
        mlp_hidden_layer(A, Wt1, b1, lane);
        mlp_hidden_layer(A, Wt2, b2, lane);

        // ---- layer 3: force[m] = dot(h2[m,:], W3) + b3 -------------------
        float sum = 0.0f;
        {
            const int koff = hi * 64;
#pragma unroll
            for (int c8 = 0; c8 < 8; ++c8) {
                const v8h hv = *(const v8h*)&A[nl * HIDDEN + koff + c8 * 8];
                const v4f w0 = *(const v4f*)(W3 + koff + c8 * 8);
                const v4f w1 = *(const v4f*)(W3 + koff + c8 * 8 + 4);
#pragma unroll
                for (int i = 0; i < 4; ++i) {
                    sum += (float)hv[i]     * w0[i];
                    sum += (float)hv[i + 4] * w1[i];
                }
            }
        }
        sRed[wave][lane] = sum;
        // in-wave LDS ordering again
        if (hi == 0 && valid) {
            const float force = sRed[wave][nl] + sRed[wave][nl + 16] + b3[0];
            atomicAdd(&out[2 * dIdx],     force * ux);
            atomicAdd(&out[2 * dIdx + 1], force * uy);
        }
    }
}

extern "C" void kernel_launch(void* const* d_in, const int* in_sizes, int n_in,
                              void* d_out, int out_size, void* d_ws, size_t ws_size,
                              hipStream_t stream) {
    // setup_inputs order: x, v, src, dst, gamma, W0, b0, W1, b1, W2, b2, W3, b3
    const float* x     = (const float*)d_in[0];
    const float* v     = (const float*)d_in[1];
    const int*   src   = (const int*)d_in[2];
    const int*   dst   = (const int*)d_in[3];
    const float* gamma = (const float*)d_in[4];
    const float* W0    = (const float*)d_in[5];
    const float* b0    = (const float*)d_in[6];
    const float* W1    = (const float*)d_in[7];
    const float* b1    = (const float*)d_in[8];
    const float* W2    = (const float*)d_in[9];
    const float* b2    = (const float*)d_in[10];
    const float* W3    = (const float*)d_in[11];
    const float* b3    = (const float*)d_in[12];

    float* out = (float*)d_out;
    const int nEdges = in_sizes[2];

    // d_ws: [0 .. 128*128) f16 Wt1, [128*128 .. 2*128*128) f16 Wt2
    _Float16* wt = (_Float16*)d_ws;

    // 1) convert + transpose hidden weights to f16 (32 KB each, hot in cache)
    wconvert_kernel<<<(HIDDEN * HIDDEN + 255) / 256, 256, 0, stream>>>(W1, W2, wt);

    // 2) out = -gamma * v
    out_init_kernel<<<(out_size + 255) / 256, 256, 0, stream>>>(v, gamma, out, out_size);

    // 3) fused edge MLP + scatter-add
    edge_force_kernel<<<EDGE_BLOCKS, BLOCK_THREADS, 0, stream>>>(
        x, src, dst, W0, b0, b1, b2, W3, b3,
        wt, wt + HIDDEN * HIDDEN, out, nEdges);
}
